// DeepAR_77240691851870
// MI455X (gfx1250) — compile-verified, matrix operand order checked by reading, not run on hardware
//
#include <hip/hip_runtime.h>
#include <hip/hip_bf16.h>
#include <cstdint>

// ---------------------------------------------------------------------------
// DeepAR LSTM forecaster for MI455X (gfx1250, wave32, WMMA).
// B=128, N=H=512, L=3, S=336 encoder steps, P=96 decoder steps.
// Strategy: bf16 WMMA (f32 accum) GEMMs; fused gate layout so each wave holds
// i/f/g/o accumulators for the same (b,h) 16x16 tile -> LSTM cell is pure VALU.
// Weights packed once to bf16, live in the 192MB L2 for all 432 steps.
// ---------------------------------------------------------------------------

typedef __attribute__((ext_vector_type(16))) __bf16 bf16x16;
typedef __attribute__((ext_vector_type(8)))  float  f32x8;

#define SEQ_S 336
#define SEQ_P 96
#define NKL   32   // K=1024 / 32 per LSTM layer-step
#define NKP   48   // K=1536 / 32 for projection

__device__ __forceinline__ unsigned short f2bf(float f) {
  unsigned int u = __float_as_uint(f);
  unsigned int r = u + 0x7FFFu + ((u >> 16) & 1u);  // round-to-nearest-even
  return (unsigned short)(r >> 16);
}
__device__ __forceinline__ float sigmoidf_(float x) { return 1.f / (1.f + __expf(-x)); }
__device__ __forceinline__ float tanhf_(float x) {
  float e = __expf(2.f * x);
  return 1.f - 2.f / (e + 1.f);  // saturates correctly at +/-1
}

// ------------------------- preprocessing kernels ---------------------------

// x[B,1,N,S] fp32 -> xt[s][b][n] bf16
__global__ void k_pack_x(const float* __restrict__ x, unsigned short* __restrict__ xt) {
  int idx = blockIdx.x * 256 + threadIdx.x;
  if (idx >= SEQ_S * 128 * 512) return;
  int n = idx & 511;
  int b = (idx >> 9) & 127;
  int s = idx >> 16;
  xt[idx] = f2bf(x[((size_t)(b * 512 + n)) * SEQ_S + s]);
}

// [Wih | Whh] -> bf16 packed [hb(32)][ks(32)][gate(4)][col(16)][kk(32)]
__global__ void k_pack_lstm(const float* __restrict__ Wih, const float* __restrict__ Whh,
                            unsigned short* __restrict__ Wp) {
  int idx = blockIdx.x * 256 + threadIdx.x;
  if (idx >= 2048 * 1024) return;
  int kk  = idx & 31;
  int col = (idx >> 5) & 15;
  int g   = (idx >> 9) & 3;
  int ks  = (idx >> 11) & 31;
  int hb  = idx >> 16;
  int k = ks * 32 + kk;
  int j = g * 512 + hb * 16 + col;
  float v = (k < 512) ? Wih[(size_t)j * 512 + k] : Whh[(size_t)j * 512 + (k - 512)];
  Wp[idx] = f2bf(v);
}

__global__ void k_pack_bias(const float* __restrict__ bih, const float* __restrict__ bhh,
                            float* __restrict__ bc) {
  int idx = blockIdx.x * 256 + threadIdx.x;
  if (idx < 2048) bc[idx] = bih[idx] + bhh[idx];
}

// Wmu[N, H*L] (feature = h*L + l) -> bf16 packed [nb(32)][ks(48)][col(16)][kk(32)]
// with K ordered layer-major (k = l*512 + h) so each 32-wide K step stays in one layer.
__global__ void k_pack_wmu(const float* __restrict__ Wmu, unsigned short* __restrict__ Wmup) {
  int idx = blockIdx.x * 256 + threadIdx.x;
  if (idx >= 32 * 48 * 512) return;
  int kk  = idx & 31;
  int col = (idx >> 5) & 15;
  int ks  = (idx >> 9) % 48;
  int nb  = idx / (48 * 512);
  int k = ks * 32 + kk;
  int l = k >> 9;
  int h = k & 511;
  int n = nb * 16 + col;
  Wmup[idx] = f2bf(Wmu[(size_t)n * 1536 + h * 3 + l]);
}

__global__ void k_zero_u16(unsigned short* p, int n) {
  int i = blockIdx.x * 256 + threadIdx.x;
  if (i < n) p[i] = 0;
}
__global__ void k_zero_f32(float* p, int n) {
  int i = blockIdx.x * 256 + threadIdx.x;
  if (i < n) p[i] = 0.f;
}

// ------------------------- LSTM layer-step kernel --------------------------
// grid = (32 h-blocks, 2 row-blocks), block = 128 threads (4 waves).
// Each wave owns a 16x16 (b,h) tile and 4 WMMA accumulators (gates i,f,g,o).
// K loop = 32 iterations of 16x16x32 bf16 WMMA, LDS double-buffered.
__global__ __launch_bounds__(128) void k_lstm_layer(
    const unsigned short* __restrict__ inp,    // [128][512] bf16 (x_t or h below)
    const unsigned short* __restrict__ hprev,  // [128][512] bf16 (read parity)
    unsigned short* __restrict__ hout,         // [128][512] bf16 (write parity)
    float* __restrict__ cst,                   // [128][512] fp32, read+write
    const unsigned short* __restrict__ Wp,     // packed layer weights
    const float* __restrict__ bias) {          // [4*512] bih+bhh
  __shared__ alignas(32) unsigned short Uld[2][64][32];      // 8 KB
  __shared__ alignas(32) unsigned short Wld[2][4][16][32];   // 8 KB
  const int hb = blockIdx.x;
  const int rb = blockIdx.y;
  const int t = threadIdx.x;
  const int wave = t >> 5;
  const int lane = t & 31;
  const int urow = t >> 1;            // 0..63
  const int useg = (t & 1) << 4;      // 0 / 16
  const int growbase = rb * 64;

  const unsigned short* wsrc = Wp + (size_t)hb * (NKL * 2048);

  f32x8 z8 = {0.f, 0.f, 0.f, 0.f, 0.f, 0.f, 0.f, 0.f};
  f32x8 acc0 = z8, acc1 = z8, acc2 = z8, acc3 = z8;

  uint4 us0, us1, wstage0, wstage1;
  {  // prologue (ks=0, k0=0 -> inp half)
    const unsigned short* up = inp + (size_t)(growbase + urow) * 512 + useg;
    us0 = *(const uint4*)up;
    us1 = *(const uint4*)(up + 8);
    const unsigned short* wp0 = wsrc + (size_t)t * 16;
    wstage0 = *(const uint4*)wp0;
    wstage1 = *(const uint4*)(wp0 + 8);
  }
  {
    uint4* ud = (uint4*)&Uld[0][urow][useg];
    ud[0] = us0; ud[1] = us1;
    uint4* wd = (uint4*)((unsigned short*)&Wld[0][0][0][0] + t * 16);
    wd[0] = wstage0; wd[1] = wstage1;
  }

  for (int ks = 0; ks < NKL; ++ks) {
    __syncthreads();
    const int cur = ks & 1;
    const bool more = (ks + 1) < NKL;
    if (more) {  // stage next tiles (overlaps with WMMA below)
      const int k0 = (ks + 1) * 32;
      const unsigned short* ubase = (k0 < 512) ? (inp + k0) : (hprev + (k0 - 512));
      const unsigned short* up = ubase + (size_t)(growbase + urow) * 512 + useg;
      us0 = *(const uint4*)up;
      us1 = *(const uint4*)(up + 8);
      const unsigned short* wp0 = wsrc + (size_t)(ks + 1) * 2048 + (size_t)t * 16;
      wstage0 = *(const uint4*)wp0;
      wstage1 = *(const uint4*)(wp0 + 8);
    }
    const int arow = wave * 16 + (lane & 15);
    const int aseg = (lane >> 4) << 4;
    bf16x16 a  = *(const bf16x16*)&Uld[cur][arow][aseg];
    bf16x16 b0 = *(const bf16x16*)&Wld[cur][0][lane & 15][aseg];
    bf16x16 b1 = *(const bf16x16*)&Wld[cur][1][lane & 15][aseg];
    bf16x16 b2 = *(const bf16x16*)&Wld[cur][2][lane & 15][aseg];
    bf16x16 b3 = *(const bf16x16*)&Wld[cur][3][lane & 15][aseg];
    acc0 = __builtin_amdgcn_wmma_f32_16x16x32_bf16(false, a, false, b0, (short)0, acc0, false, false);
    acc1 = __builtin_amdgcn_wmma_f32_16x16x32_bf16(false, a, false, b1, (short)0, acc1, false, false);
    acc2 = __builtin_amdgcn_wmma_f32_16x16x32_bf16(false, a, false, b2, (short)0, acc2, false, false);
    acc3 = __builtin_amdgcn_wmma_f32_16x16x32_bf16(false, a, false, b3, (short)0, acc3, false, false);
    __syncthreads();
    if (more) {
      const int nxt = cur ^ 1;
      uint4* ud = (uint4*)&Uld[nxt][urow][useg];
      ud[0] = us0; ud[1] = us1;
      uint4* wd = (uint4*)((unsigned short*)&Wld[nxt][0][0][0] + t * 16);
      wd[0] = wstage0; wd[1] = wstage1;
    }
  }

  // Fused LSTM cell: accumulators of all 4 gates share the (b,h) lane mapping.
  const int hcol = hb * 16 + (lane & 15);
  const float bi = bias[0 * 512 + hcol];
  const float bff = bias[1 * 512 + hcol];
  const float bg = bias[2 * 512 + hcol];
  const float bo = bias[3 * 512 + hcol];
  const int rbase = growbase + wave * 16 + ((lane >> 4) << 3);
#pragma unroll
  for (int r = 0; r < 8; ++r) {
    const size_t idx = (size_t)(rbase + r) * 512 + hcol;
    const float gi = acc0[r] + bi;
    const float gf = acc1[r] + bff;
    const float gg = acc2[r] + bg;
    const float go = acc3[r] + bo;
    const float cn = sigmoidf_(gf) * cst[idx] + sigmoidf_(gi) * tanhf_(gg);
    cst[idx] = cn;
    hout[idx] = f2bf(sigmoidf_(go) * tanhf_(cn));
  }
}

// ------------------------- projection kernel -------------------------------
// mu[b,n] = hp[b,:] @ Wmu^T + bmu ; emit clamp(mu) to d_out[:, :, p] and mu_bf.
__global__ __launch_bounds__(128) void k_proj(
    const unsigned short* __restrict__ hbf,  // [3][2][128][512] bf16
    int par,
    const unsigned short* __restrict__ Wmup,
    const float* __restrict__ bmu,
    float* __restrict__ out, int pidx,
    unsigned short* __restrict__ mubf) {
  __shared__ alignas(32) unsigned short Uld[2][64][32];  // 8 KB
  __shared__ alignas(32) unsigned short Wld[2][16][32];  // 2 KB
  const int nb = blockIdx.x;
  const int rb = blockIdx.y;
  const int t = threadIdx.x;
  const int wave = t >> 5;
  const int lane = t & 31;
  const int urow = t >> 1;
  const int useg = (t & 1) << 4;
  const int growbase = rb * 64;

  const unsigned short* wsrc = Wmup + (size_t)nb * (NKP * 512);

  f32x8 acc = {0.f, 0.f, 0.f, 0.f, 0.f, 0.f, 0.f, 0.f};

  uint4 us0, us1; uint2 wstage;
  {  // prologue (ks=0 -> layer 0)
    const unsigned short* hp = hbf + (size_t)par * (128 * 512);
    const unsigned short* up = hp + (size_t)(growbase + urow) * 512 + useg;
    us0 = *(const uint4*)up;
    us1 = *(const uint4*)(up + 8);
    wstage = *(const uint2*)(wsrc + (size_t)t * 4);
  }
  {
    uint4* ud = (uint4*)&Uld[0][urow][useg];
    ud[0] = us0; ud[1] = us1;
    *(uint2*)((unsigned short*)&Wld[0][0][0] + t * 4) = wstage;
  }

  for (int ks = 0; ks < NKP; ++ks) {
    __syncthreads();
    const int cur = ks & 1;
    const bool more = (ks + 1) < NKP;
    if (more) {
      const int ksn = ks + 1;
      const int l = ksn >> 4;               // 16 k-steps per layer
      const int k0 = (ksn & 15) * 32;
      const unsigned short* hp = hbf + (size_t)(l * 2 + par) * (128 * 512);
      const unsigned short* up = hp + (size_t)(growbase + urow) * 512 + k0 + useg;
      us0 = *(const uint4*)up;
      us1 = *(const uint4*)(up + 8);
      wstage = *(const uint2*)(wsrc + (size_t)ksn * 512 + (size_t)t * 4);
    }
    const int arow = wave * 16 + (lane & 15);
    const int aseg = (lane >> 4) << 4;
    bf16x16 a = *(const bf16x16*)&Uld[cur][arow][aseg];
    bf16x16 b = *(const bf16x16*)&Wld[cur][lane & 15][aseg];
    acc = __builtin_amdgcn_wmma_f32_16x16x32_bf16(false, a, false, b, (short)0, acc, false, false);
    __syncthreads();
    if (more) {
      const int nxt = cur ^ 1;
      uint4* ud = (uint4*)&Uld[nxt][urow][useg];
      ud[0] = us0; ud[1] = us1;
      *(uint2*)((unsigned short*)&Wld[nxt][0][0] + t * 4) = wstage;
    }
  }

  const int ncol = nb * 16 + (lane & 15);
  const float bn = bmu[ncol];
  const int rbase = growbase + wave * 16 + ((lane >> 4) << 3);
#pragma unroll
  for (int r = 0; r < 8; ++r) {
    float v = fmaxf(acc[r] + bn, 1e-9f);   // clamp(min=1e-9): emitted prediction
    const int b = rbase + r;
    out[((size_t)b * 512 + ncol) * 96 + pidx] = v;
    mubf[(size_t)b * 512 + ncol] = f2bf(v);
  }
}

// ------------------------------ host driver --------------------------------

extern "C" void kernel_launch(void* const* d_in, const int* in_sizes, int n_in,
                              void* d_out, int out_size, void* d_ws, size_t ws_size,
                              hipStream_t stream) {
  (void)in_sizes; (void)n_in; (void)out_size; (void)ws_size;
  const float* x = (const float*)d_in[0];
  const float* Wih[3] = {(const float*)d_in[1], (const float*)d_in[5], (const float*)d_in[9]};
  const float* Whh[3] = {(const float*)d_in[2], (const float*)d_in[6], (const float*)d_in[10]};
  const float* bih[3] = {(const float*)d_in[3], (const float*)d_in[7], (const float*)d_in[11]};
  const float* bhh[3] = {(const float*)d_in[4], (const float*)d_in[8], (const float*)d_in[12]};
  const float* Wmu = (const float*)d_in[13];
  const float* bmu = (const float*)d_in[14];
  float* out = (float*)d_out;

  char* ws = (char*)d_ws;
  size_t off = 0;
  unsigned short* xt = (unsigned short*)(ws + off);   off += (size_t)SEQ_S * 128 * 512 * 2;
  unsigned short* Wp = (unsigned short*)(ws + off);   off += (size_t)3 * 2048 * 1024 * 2;
  float* biasc = (float*)(ws + off);                  off += (size_t)3 * 2048 * 4;
  unsigned short* Wmup = (unsigned short*)(ws + off); off += (size_t)32 * 48 * 512 * 2;
  unsigned short* hbf = (unsigned short*)(ws + off);  off += (size_t)3 * 2 * 128 * 512 * 2;
  float* cst = (float*)(ws + off);                    off += (size_t)3 * 128 * 512 * 4;
  unsigned short* mubf = (unsigned short*)(ws + off); off += (size_t)128 * 512 * 2;

  // --- one-time (per call) bf16 packing + state init (deterministic) ---
  k_pack_x<<<(SEQ_S * 128 * 512 + 255) / 256, 256, 0, stream>>>(x, xt);
  for (int l = 0; l < 3; ++l) {
    k_pack_lstm<<<(2048 * 1024) / 256, 256, 0, stream>>>(Wih[l], Whh[l],
                                                         Wp + (size_t)l * 2048 * 1024);
    k_pack_bias<<<8, 256, 0, stream>>>(bih[l], bhh[l], biasc + l * 2048);
  }
  k_pack_wmu<<<(32 * 48 * 512) / 256, 256, 0, stream>>>(Wmu, Wmup);
  k_zero_u16<<<(3 * 2 * 128 * 512) / 256, 256, 0, stream>>>(hbf, 3 * 2 * 128 * 512);
  k_zero_f32<<<(3 * 128 * 512) / 256, 256, 0, stream>>>(cst, 3 * 128 * 512);

  dim3 gridL(32, 2), blockL(128);
  int step = 0;
  auto launch_step = [&](const unsigned short* in0) {
    const int rp = step & 1, wp = 1 - rp;  // h parity ping-pong per step
    for (int l = 0; l < 3; ++l) {
      const unsigned short* inp =
          (l == 0) ? in0 : (hbf + (size_t)((l - 1) * 2 + wp) * (128 * 512));
      k_lstm_layer<<<gridL, blockL, 0, stream>>>(
          inp,
          hbf + (size_t)(l * 2 + rp) * (128 * 512),
          hbf + (size_t)(l * 2 + wp) * (128 * 512),
          cst + (size_t)l * (128 * 512),
          Wp + (size_t)l * (2048 * 1024),
          biasc + l * 2048);
    }
    step++;
  };

  // encoder: 336 steps
  for (int s = 0; s < SEQ_S; ++s) launch_step(xt + (size_t)s * (128 * 512));
  // mu_0 -> pred[0] + first decoder input
  k_proj<<<gridL, blockL, 0, stream>>>(hbf, step & 1, Wmup, bmu, out, 0, mubf);
  // decoder: 95 LSTM steps, each followed by projection emitting pred[p]
  for (int p = 1; p < SEQ_P; ++p) {
    launch_step(mubf);
    k_proj<<<gridL, blockL, 0, stream>>>(hbf, step & 1, Wmup, bmu, out, p, mubf);
  }
}
